// HyperAttention_72172630442646
// MI455X (gfx1250) — matrix operand order repaired and checked by reading, not compiled, not run on hardware
//
#include <hip/hip_runtime.h>
#include <math.h>

typedef __attribute__((ext_vector_type(16))) _Float16 v16h;
typedef __attribute__((ext_vector_type(8)))  _Float16 v8h;
typedef __attribute__((ext_vector_type(8)))  float    v8f;

#define WAVE 32

// ---------------------------------------------------------------------------
// CDNA5 async global->LDS copy (ASYNCcnt path, ISA §15.18.3 op 98).
// Generic shared-pointer low 32 bits == LDS byte offset (ISA §10.2).
// ---------------------------------------------------------------------------
__device__ __forceinline__ unsigned lds_off(const void* p) {
  return (unsigned)(uintptr_t)p;
}
__device__ __forceinline__ void async_copy_b128(unsigned dst_lds, const void* src) {
  asm volatile("global_load_async_to_lds_b128 %0, %1, off"
               :: "v"(dst_lds), "v"(src) : "memory");
}
__device__ __forceinline__ void wait_async0() {
  asm volatile("s_wait_asynccnt 0x0" ::: "memory");
}

// ---------------------------------------------------------------------------
// WMMA fragment loaders (layouts per CDNA5 ISA §7.12.2, wave32)
// ---------------------------------------------------------------------------
__device__ __forceinline__ v16h load_a_frag(const _Float16* tile, int ldr, int k0, int lane) {
  int m  = lane & 15;
  int ko = k0 + ((lane >> 4) << 3);
  const _Float16* p = tile + m * ldr + ko;
  v8h lo = *(const v8h*)p;
  v8h hi = *(const v8h*)(p + 16);
  v16h r;
#pragma unroll
  for (int i = 0; i < 8; ++i) { r[i] = lo[i]; r[i + 8] = hi[i]; }
  return r;
}

__device__ __forceinline__ v16h load_b_frag(const _Float16* tile, int ldr, int nrow0, int k0, int lane) {
  int n  = lane & 15;
  int ko = k0 + ((lane >> 4) << 3);
  const _Float16* p = tile + (nrow0 + n) * ldr + ko;
  v8h lo = *(const v8h*)p;
  v8h hi = *(const v8h*)(p + 16);
  v16h r;
#pragma unroll
  for (int i = 0; i < 8; ++i) { r[i] = lo[i]; r[i + 8] = hi[i]; }
  return r;
}

// ---------------------------------------------------------------------------
// Flash-attention tile kernel: one wave per (32-row q-tile, head).
// Flat row of all tensors = pos*H + hh (layout-preserving reshapes).
// F16IN: inputs pre-converted to f16 (Q pre-scaled); LDS filled with async copies.
// ---------------------------------------------------------------------------
template <bool F16IN>
__global__ void __launch_bounds__(32)
attn_kernel(const float* __restrict__ Qb, const float* __restrict__ Kb,
            const float* __restrict__ Vb,
            const _Float16* __restrict__ Qh, const _Float16* __restrict__ Kh,
            const _Float16* __restrict__ Vh,
            float* __restrict__ O, float* __restrict__ LSE,
            int H, int q0, int nk, int k0,
            const int* __restrict__ q_idx, const int* __restrict__ k_idx,
            int causal, int block_diag, int obase, float scale)
{
  __shared__ __attribute__((aligned(32))) _Float16 sQ[32 * 64];
  __shared__ __attribute__((aligned(32))) _Float16 sK[32 * 64];
  __shared__ __attribute__((aligned(32))) _Float16 sVr[32 * 64];   // used by F16IN only
  __shared__ __attribute__((aligned(32))) _Float16 sVt[64 * 32];
  __shared__ __attribute__((aligned(32))) _Float16 sP[32 * 32];

  const int lane = threadIdx.x;
  const int hh = blockIdx.x % H;
  const int qt = blockIdx.x / H;

  // ---- Q tile (32 x 64)
  if constexpr (F16IN) {
    for (int c = lane; c < 32 * 8; c += WAVE) {          // 16B segments
      int m = c >> 3, seg = c & 7;
      int qp = q_idx ? q_idx[(qt * 32 + m) * H + hh] : (q0 + qt * 32 + m);
      async_copy_b128(lds_off(sQ + m * 64 + seg * 8),
                      Qh + ((size_t)qp * H + hh) * 64 + seg * 8);
    }
    wait_async0();
  } else {
    for (int e = lane; e < 32 * 64; e += WAVE) {
      int m = e >> 6, d = e & 63;
      int qp = q_idx ? q_idx[(qt * 32 + m) * H + hh] : (q0 + qt * 32 + m);
      sQ[e] = (_Float16)(Qb[((size_t)qp * H + hh) * 64 + d] * scale);
    }
  }
  __syncthreads();
  v16h aQ[2][2];
#pragma unroll
  for (int h2 = 0; h2 < 2; ++h2) {
    aQ[h2][0] = load_a_frag(sQ + h2 * 16 * 64, 64, 0, lane);
    aQ[h2][1] = load_a_frag(sQ + h2 * 16 * 64, 64, 32, lane);
  }

  v8f o[2][4];
  float m_[2][8], l_[2][8];
#pragma unroll
  for (int h2 = 0; h2 < 2; ++h2)
#pragma unroll
    for (int r = 0; r < 8; ++r) { o[h2][r >> 1] = {}; m_[h2][r] = -1e30f; l_[h2][r] = 0.f; }
#pragma unroll
  for (int h2 = 0; h2 < 2; ++h2)
#pragma unroll
    for (int t = 0; t < 4; ++t) o[h2][t] = v8f{};

  const int kwin0 = block_diag ? (qt / 8) * 256 : 0;
  int nch = block_diag ? 8 : (nk >> 5);
  if (causal) { int mx = qt + 1; if (mx < nch) nch = mx; }

  float alph[2][8];

  for (int kc = 0; kc < nch; ++kc) {
    const int kbase = kwin0 + kc * 32;
    __syncthreads();
    // ---- K chunk (32x64) and V chunk (transposed into sVt)
    if constexpr (F16IN) {
      for (int c = lane; c < 32 * 8; c += WAVE) {
        int j = c >> 3, seg = c & 7;
        int kp = k_idx ? k_idx[(kbase + j) * H + hh] : (k0 + kbase + j);
        size_t g = ((size_t)kp * H + hh) * 64 + seg * 8;
        async_copy_b128(lds_off(sK + j * 64 + seg * 8), Kh + g);
        async_copy_b128(lds_off(sVr + j * 64 + seg * 8), Vh + g);
      }
      wait_async0();
      __syncthreads();
      for (int c = lane; c < 32 * 8; c += WAVE) {        // LDS->LDS transpose
        int j = c >> 3, seg = c & 7;
        v8h x = *(const v8h*)(sVr + j * 64 + seg * 8);
#pragma unroll
        for (int i = 0; i < 8; ++i) sVt[(seg * 8 + i) * 32 + j] = x[i];
      }
    } else {
      for (int e = lane; e < 32 * 64; e += WAVE) {
        int j = e >> 6, d = e & 63;
        int kp = k_idx ? k_idx[(kbase + j) * H + hh] : (k0 + kbase + j);
        size_t g = ((size_t)kp * H + hh) * 64 + d;
        sK[e] = (_Float16)Kb[g];
        sVt[d * 32 + j] = (_Float16)Vb[g];
      }
    }
    __syncthreads();

    // ---- S = Q K^T, online softmax, P -> LDS (per 16-row half)
#pragma unroll
    for (int h2 = 0; h2 < 2; ++h2) {
      v8f s0 = {}, s1 = {};
      s0 = __builtin_amdgcn_wmma_f32_16x16x32_f16(false, aQ[h2][0], false, load_b_frag(sK, 64, 0, 0, lane),  (short)0, s0, false, false);
      s0 = __builtin_amdgcn_wmma_f32_16x16x32_f16(false, aQ[h2][1], false, load_b_frag(sK, 64, 0, 32, lane), (short)0, s0, false, false);
      s1 = __builtin_amdgcn_wmma_f32_16x16x32_f16(false, aQ[h2][0], false, load_b_frag(sK, 64, 16, 0, lane),  (short)0, s1, false, false);
      s1 = __builtin_amdgcn_wmma_f32_16x16x32_f16(false, aQ[h2][1], false, load_b_frag(sK, 64, 16, 32, lane), (short)0, s1, false, false);

      const int rowb = qt * 32 + h2 * 16 + ((lane >> 4) << 3);
      if (causal && (kbase + 31 > qt * 32 + h2 * 16)) {
        int col0 = kbase + (lane & 15);
#pragma unroll
        for (int r = 0; r < 8; ++r) {
          if (col0      > rowb + r) s0[r] = -3.0e38f;
          if (col0 + 16 > rowb + r) s1[r] = -3.0e38f;
        }
      }

#pragma unroll
      for (int r = 0; r < 8; ++r) {
        float x = fmaxf(s0[r], s1[r]);
#pragma unroll
        for (int off = 1; off < 16; off <<= 1) x = fmaxf(x, __shfl_xor(x, off, WAVE));
        float mn = fmaxf(m_[h2][r], x);
        float a  = __expf(m_[h2][r] - mn);
        float p0 = __expf(s0[r] - mn);
        float p1 = __expf(s1[r] - mn);
        float y  = p0 + p1;
#pragma unroll
        for (int off = 1; off < 16; off <<= 1) y += __shfl_xor(y, off, WAVE);
        l_[h2][r] = l_[h2][r] * a + y;
        m_[h2][r] = mn; alph[h2][r] = a;
        s0[r] = p0; s1[r] = p1;
      }
      {
        int rr = h2 * 16 + ((lane >> 4) << 3);
        int cc = lane & 15;
#pragma unroll
        for (int r = 0; r < 8; ++r) {
          sP[(r + rr) * 32 + cc]      = (_Float16)s0[r];
          sP[(r + rr) * 32 + 16 + cc] = (_Float16)s1[r];
        }
      }
    }
    __syncthreads();

    // ---- O += P V
#pragma unroll
    for (int h2 = 0; h2 < 2; ++h2) {
      v16h aP = load_a_frag(sP + h2 * 16 * 32, 32, 0, lane);
#pragma unroll
      for (int t = 0; t < 4; ++t)
#pragma unroll
        for (int r = 0; r < 8; ++r) o[h2][t][r] *= alph[h2][r];
      o[h2][0] = __builtin_amdgcn_wmma_f32_16x16x32_f16(false, aP, false, load_b_frag(sVt, 32, 0,  0, lane), (short)0, o[h2][0], false, false);
      o[h2][1] = __builtin_amdgcn_wmma_f32_16x16x32_f16(false, aP, false, load_b_frag(sVt, 32, 16, 0, lane), (short)0, o[h2][1], false, false);
      o[h2][2] = __builtin_amdgcn_wmma_f32_16x16x32_f16(false, aP, false, load_b_frag(sVt, 32, 32, 0, lane), (short)0, o[h2][2], false, false);
      o[h2][3] = __builtin_amdgcn_wmma_f32_16x16x32_f16(false, aP, false, load_b_frag(sVt, 32, 48, 0, lane), (short)0, o[h2][3], false, false);
    }
  }

#pragma unroll
  for (int h2 = 0; h2 < 2; ++h2)
#pragma unroll
    for (int r = 0; r < 8; ++r) {
      int M = h2 * 16 + r + ((lane >> 4) << 3);
      int qp = q_idx ? q_idx[(qt * 32 + M) * H + hh] : (q0 + qt * 32 + M);
      size_t orow = (size_t)qp * H + hh - (size_t)obase;
      float inv = 1.0f / l_[h2][r];
      float* op = O + orow * 64 + (lane & 15);
      op[0]  = o[h2][0][r] * inv;
      op[16] = o[h2][1][r] * inv;
      op[32] = o[h2][2][r] * inv;
      op[48] = o[h2][3][r] * inv;
      if ((lane & 15) == 0) LSE[orow] = m_[h2][r] + __logf(l_[h2][r]);
    }
}

// ---------------------------------------------------------------------------
// f32 -> f16 conversion (optionally scaled) for the fast input path
// ---------------------------------------------------------------------------
__global__ void cvt_kernel(const float* __restrict__ X, _Float16* __restrict__ Y,
                           float mul, unsigned n)
{
  unsigned t = blockIdx.x * 256 + threadIdx.x;
  if (t < n) Y[t] = (_Float16)(X[t] * mul);
}

// ---------------------------------------------------------------------------
// LSH hash: PERM is Gray-code order -> bucket = b ^ (b>>1)
// ---------------------------------------------------------------------------
__global__ void hash_kernel(const float* __restrict__ X, const float* __restrict__ proj,
                            int* __restrict__ out, int H, int n, int i0)
{
  int t = blockIdx.x * 256 + threadIdx.x;
  if (t >= n * H) return;
  int pos = t / H, hh = t % H;
  const float* x = X + ((size_t)(i0 + pos) * H + hh) * 64;
  float acc[8] = {0, 0, 0, 0, 0, 0, 0, 0};
  for (int d = 0; d < 64; ++d) {
    float xv = x[d];
#pragma unroll
    for (int r = 0; r < 8; ++r) acc[r] += xv * proj[d * 8 + r];
  }
  int b = 0;
#pragma unroll
  for (int r = 0; r < 8; ++r) b |= (acc[r] > 0.f) << r;
  out[pos * H + hh] = b ^ (b >> 1);
}

// ---------------------------------------------------------------------------
// Stable counting argsort over 256 buckets, one block (256 thr) per head.
// ---------------------------------------------------------------------------
__global__ void argsort_kernel(const int* __restrict__ hash, int* __restrict__ idx,
                               int H, int n, int i0)
{
  __shared__ int cnt[256];
  __shared__ int off[256];
  __shared__ int bk[256];
  __shared__ int ccnt[256];
  int hh = blockIdx.x;
  int t = threadIdx.x;
  cnt[t] = 0;
  __syncthreads();
  for (int e = t; e < n; e += 256) atomicAdd(&cnt[hash[e * H + hh]], 1);
  __syncthreads();
  if (t == 0) { int s = 0; for (int i = 0; i < 256; ++i) { off[i] = s; s += cnt[i]; } }
  __syncthreads();
  int nch = n / 256;
  for (int c = 0; c < nch; ++c) {
    int e = c * 256 + t;
    int b = hash[e * H + hh];
    bk[t] = b;
    __syncthreads();
    int rank = 0;
    for (int j = 0; j < t; ++j) rank += (bk[j] == b);
    idx[(off[b] + rank) * H + hh] = i0 + e;
    ccnt[t] = 0;
    __syncthreads();
    atomicAdd(&ccnt[b], 1);
    __syncthreads();
    off[t] += ccnt[t];
    __syncthreads();
  }
}

// ---------------------------------------------------------------------------
// threefry2x32, 16 rounds (JAX PRNG); sampled = bits % range
// ---------------------------------------------------------------------------
__device__ __forceinline__ unsigned tf_rotl(unsigned v, int n) { return (v << n) | (v >> (32 - n)); }

__device__ void threefry2x32(unsigned k0, unsigned k1, unsigned c0, unsigned c1,
                             unsigned* r0, unsigned* r1)
{
  const unsigned k2 = 0x1BD11BDAu ^ k0 ^ k1;
  unsigned x0 = c0 + k0, x1 = c1 + k1;
  const int RA[4] = {13, 15, 26, 6}, RB[4] = {17, 29, 16, 24};
#pragma unroll
  for (int i = 0; i < 4; ++i) { x0 += x1; x1 = tf_rotl(x1, RA[i]); x1 ^= x0; }
  x0 += k1; x1 += k2 + 1u;
#pragma unroll
  for (int i = 0; i < 4; ++i) { x0 += x1; x1 = tf_rotl(x1, RB[i]); x1 ^= x0; }
  x0 += k2; x1 += k0 + 2u;
#pragma unroll
  for (int i = 0; i < 4; ++i) { x0 += x1; x1 = tf_rotl(x1, RA[i]); x1 ^= x0; }
  x0 += k0; x1 += k1 + 3u;
#pragma unroll
  for (int i = 0; i < 4; ++i) { x0 += x1; x1 = tf_rotl(x1, RB[i]); x1 ^= x0; }
  x0 += k1; x1 += k2 + 4u;
  *r0 = x0; *r1 = x1;
}

__global__ void sample_kernel(int* __restrict__ out, int H, int range, int tag, int count)
{
  int t = blockIdx.x * 256 + threadIdx.x;
  if (t >= count) return;
  unsigned f0, f1, r0, r1;
  threefry2x32(0u, 42u, 0u, (unsigned)tag, &f0, &f1);   // fold_in(key(42), tag)
  threefry2x32(f0, f1, 0u, (unsigned)t, &r0, &r1);
  out[t] = (int)(r0 % (unsigned)range);
}

// ---------------------------------------------------------------------------
// online-softmax merge: (O1,L1) <- add_attn((O1,L1), (O2, L2 + l2off))
// block of 256 = 4 whole rows -> race-free read-then-write of L1.
// ---------------------------------------------------------------------------
__global__ void merge_kernel(float* __restrict__ O1, float* __restrict__ L1,
                             const float* __restrict__ O2, const float* __restrict__ L2,
                             float l2off)
{
  int t = blockIdx.x * 256 + threadIdx.x;
  int row = t >> 6, d = t & 63;
  float l1 = L1[row], l2 = L2[row] + l2off;
  float mx = fmaxf(l1, l2);
  float lse = mx + __logf(__expf(l1 - mx) + __expf(l2 - mx));
  float c = 1.0f / (1.0f + __expf(l2 - l1));       // sigmoid(l1 - l2)
  float o = c * O1[t] + (1.0f - c) * O2[t];
  __syncthreads();
  O1[t] = o;
  if (d == 0) L1[row] = lse;
}

// ---------------------------------------------------------------------------
extern "C" void kernel_launch(void* const* d_in, const int* in_sizes, int n_in,
                              void* d_out, int out_size, void* d_ws, size_t ws_size,
                              hipStream_t stream)
{
  (void)in_sizes; (void)n_in; (void)out_size;
  const float* Q = (const float*)d_in[0];
  const float* K = (const float*)d_in[1];
  const float* V = (const float*)d_in[2];
  const float* P = (const float*)d_in[3];
  float* OUT = (float*)d_out;                 // running a_bd, flat (131072 x 64)
  const float scale = 0.125f;                 // 64^-0.5

  const size_t R = 131072;                    // total flat rows (8192*16)
  const size_t UH = R / 2;                    // upper-half rows = 65536
  float* LSE   = (float*)d_ws;                // R
  float* TO    = LSE + R;                     // UH*64
  float* TL    = TO + UH * 64;                // UH
  float* T2O   = TL + UH;                     // UH*64
  float* T2L   = T2O + UH * 64;               // UH
  int*   hq    = (int*)(T2L + UH);            // 65536
  int*   hk    = hq + 65536;
  int*   qidx  = hk + 65536;
  int*   kidx  = qidx + 65536;
  int*   samp  = kidx + 65536;                // 8192

  const size_t base_bytes = (size_t)(samp + 8192 - (int*)d_ws) * 4;
  const size_t f16_bytes  = 3 * R * 64 * sizeof(_Float16);
  const bool use_f16 = (ws_size >= base_bytes + f16_bytes);
  _Float16* Qh = (_Float16*)((char*)d_ws + base_bytes);
  _Float16* Kh = Qh + R * 64;
  _Float16* Vh = Kh + R * 64;

  if (use_f16) {
    unsigned n = (unsigned)(R * 64);
    cvt_kernel<<<n / 256, 256, 0, stream>>>(Q, Qh, scale, n);
    cvt_kernel<<<n / 256, 256, 0, stream>>>(K, Kh, 1.0f, n);
    cvt_kernel<<<n / 256, 256, 0, stream>>>(V, Vh, 1.0f, n);
  }

  auto launch_attn = [&](int grid, float* O_, float* L_, int H, int q0, int nk, int k0,
                         const int* qi, const int* ki, int causal, int bd, int obase) {
    if (use_f16)
      attn_kernel<true><<<grid, 32, 0, stream>>>(Q, K, V, Qh, Kh, Vh, O_, L_,
          H, q0, nk, k0, qi, ki, causal, bd, obase, scale);
    else
      attn_kernel<false><<<grid, 32, 0, stream>>>(Q, K, V, nullptr, nullptr, nullptr, O_, L_,
          H, q0, nk, k0, qi, ki, causal, bd, obase, scale);
  };

  const int mergeGrid = (int)(UH * 64 / 256);

  // ---- level 3 base: causal, H=128, N=1024 (covers all rows)
  launch_attn((1024 / 32) * 128, OUT, LSE, 128, 0, 1024, 0, nullptr, nullptr, 1, 0, 0);

  // ---- level 2 off-diag: H=64, q [1024,2048) x k [0,1024), exact non-causal
  launch_attn((1024 / 32) * 64, TO, TL, 64, 1024, 1024, 0, nullptr, nullptr, 0, 0, (int)UH);
  merge_kernel<<<mergeGrid, 256, 0, stream>>>(OUT + UH * 64, LSE + UH, TO, TL, 0.0f);

  // ---- level 1 off-diag: H=32, q [2048,4096) x k [0,2048) -> LSH path (tag 101)
  hash_kernel<<<(2048 * 32) / 256, 256, 0, stream>>>(Q, P, hq, 32, 2048, 2048);
  hash_kernel<<<(2048 * 32) / 256, 256, 0, stream>>>(K, P, hk, 32, 2048, 0);
  argsort_kernel<<<32, 256, 0, stream>>>(hq, qidx, 32, 2048, 2048);
  argsort_kernel<<<32, 256, 0, stream>>>(hk, kidx, 32, 2048, 0);
  launch_attn((2048 / 32) * 32, TO, TL, 32, 0, 0, 0, qidx, kidx, 0, 1, (int)UH);
  sample_kernel<<<(256 * 32) / 256, 256, 0, stream>>>(samp, 32, 2048, 101, 256 * 32);
  launch_attn((2048 / 32) * 32, T2O, T2L, 32, 2048, 256, 0, nullptr, samp, 0, 0, (int)UH);
  merge_kernel<<<mergeGrid, 256, 0, stream>>>(TO, TL, T2O, T2L, logf(8.0f));
  merge_kernel<<<mergeGrid, 256, 0, stream>>>(OUT + UH * 64, LSE + UH, TO, TL, 0.0f);

  // ---- level 0 off-diag: H=16, q [4096,8192) x k [0,4096) -> LSH path (tag 100)
  hash_kernel<<<(4096 * 16) / 256, 256, 0, stream>>>(Q, P, hq, 16, 4096, 4096);
  hash_kernel<<<(4096 * 16) / 256, 256, 0, stream>>>(K, P, hk, 16, 4096, 0);
  argsort_kernel<<<16, 256, 0, stream>>>(hq, qidx, 16, 4096, 4096);
  argsort_kernel<<<16, 256, 0, stream>>>(hk, kidx, 16, 4096, 0);
  launch_attn((4096 / 32) * 16, TO, TL, 16, 0, 0, 0, qidx, kidx, 0, 1, (int)UH);
  sample_kernel<<<(256 * 16) / 256, 256, 0, stream>>>(samp, 16, 4096, 100, 256 * 16);
  launch_attn((4096 / 32) * 16, T2O, T2L, 16, 4096, 256, 0, nullptr, samp, 0, 0, (int)UH);
  merge_kernel<<<mergeGrid, 256, 0, stream>>>(TO, TL, T2O, T2L, logf(16.0f));
  merge_kernel<<<mergeGrid, 256, 0, stream>>>(OUT + UH * 64, LSE + UH, TO, TL, 0.0f);
}
